// LGCL_Encoder_27676769255726
// MI455X (gfx1250) — compile-verified, matrix operand order checked by reading, not run on hardware
//
#include <hip/hip_runtime.h>
#include <hip/hip_bf16.h>

#define N_U   100000
#define N_I   100000
#define NTOT  (N_U + N_I)
#define D     64
#define Q     128
#define NNZ   3200000
#define LAYERS 3

#define KSTEPS_SM (N_U / 32)            /* 3125 k-blocks of 32 */
#define SM_SPLITK 125                   /* blocks per matmul    */
#define SM_STEPS  (KSTEPS_SM / SM_SPLITK) /* 25 k-blocks per block */

typedef __attribute__((ext_vector_type(16))) __bf16 v16bf;
typedef __attribute__((ext_vector_type(8)))  __bf16 v8bf;
typedef __attribute__((ext_vector_type(8)))  float  v8f;

// ---------------------------------------------------------------------------
// Fragment helpers (wave32 WMMA 16x16x32 bf16)
// A (16x32, MxK) 16-bit layout (ISA 7.12.2): lane<16 -> M=lane, K in {0..7}U{16..23}
//                                            lane>=16 -> M=lane-16, K in {8..15}U{24..31}
// ---------------------------------------------------------------------------
__device__ __forceinline__ v16bf load_a16x32(const __bf16* __restrict__ A, int lda,
                                             int m0, int k0, int lane) {
  const int half = lane >> 4;
  const int m    = lane & 15;
  const __bf16* p = A + (size_t)(m0 + m) * lda + k0 + half * 8;
  v8bf lo = *(const v8bf*)p;         // K = half*8 + 0..7
  v8bf hi = *(const v8bf*)(p + 16);  // K = 16 + half*8 + 0..7
  v16bf r;
#pragma unroll
  for (int e = 0; e < 8; ++e) { r[e] = lo[e]; r[8 + e] = hi[e]; }
  return r;
}

// B (32x16, KxN) stored pre-swizzled: for k-block kb and n-tile nt, each lane's
// 16 fragment elements are contiguous (32B). A full kb slab (4 n-tiles) is a
// contiguous 4KB run, which makes LDS staging a straight block copy.
__device__ __forceinline__ v16bf load_b_frag_ptr(const __bf16* p) {
  v8bf lo = *(const v8bf*)p;
  v8bf hi = *(const v8bf*)(p + 8);
  v16bf r;
#pragma unroll
  for (int e = 0; e < 8; ++e) { r[e] = lo[e]; r[8 + e] = hi[e]; }
  return r;
}

__device__ __forceinline__ size_t b_swizzle_idx(int r, int d_) {
  const int kr   = r & 31;
  const int half = (kr >> 3) & 1;
  const int e    = (kr & 7) | ((kr >> 4) << 3);
  const int lane = (half << 4) | (d_ & 15);
  return ((((size_t)(r >> 5)) * (D / 16) + (d_ >> 4)) * 32 + lane) * 16 + e;
}

// ---------------------------------------------------------------------------
// Kernels
// ---------------------------------------------------------------------------
__global__ void k_f32_to_bf16(const float* __restrict__ x, __bf16* __restrict__ y, int n) {
  int i = blockIdx.x * blockDim.x + threadIdx.x;
  const int stride = gridDim.x * blockDim.x;
  for (; i < n; i += stride) y[i] = (__bf16)x[i];
}

// COO SpMM scatter: one wave per edge; edge metadata is wave-uniform -> force
// onto the scalar path; lanes stream a coalesced float2 gather + f32 atomics.
__global__ void k_spmm(const int* __restrict__ row, const int* __restrict__ col,
                       const float* __restrict__ vals, const float* __restrict__ x,
                       float* __restrict__ y) {
  const int lane = threadIdx.x & 31;
  int wid        = (blockIdx.x * blockDim.x + threadIdx.x) >> 5;
  const int nw   = (gridDim.x * blockDim.x) >> 5;
  for (int e = wid; e < NNZ; e += nw) {
    const int eu = __builtin_amdgcn_readfirstlane(e);
    const int r  = __builtin_amdgcn_readfirstlane(row[eu]);
    const int c  = __builtin_amdgcn_readfirstlane(col[eu]);
    const float v = vals[eu];
    const float2 xv = *(const float2*)(x + (size_t)c * D + lane * 2);
    float* yp = y + (size_t)r * D + lane * 2;
    atomicAdd(yp,     v * xv.x);
    atomicAdd(yp + 1, v * xv.y);
  }
}

// mean accumulation: fully coalesced float4 path into the output segments.
__global__ void k_acc(const float* __restrict__ ego, float* __restrict__ out) {
  const size_t NUD4 = (size_t)N_U * D / 4;
  const float4* e4 = (const float4*)ego;
  float4* o4 = (float4*)out;
  size_t i = (size_t)blockIdx.x * blockDim.x + threadIdx.x;
  const size_t stride = (size_t)gridDim.x * blockDim.x;
  for (; i < (size_t)NTOT * D / 4; i += stride) {
    const size_t oi = (i < NUD4) ? i : i + NUD4;  // mean_u seg0, mean_i seg2
    float4 a = o4[oi], b = e4[i];
    a.x += b.x; a.y += b.y; a.z += b.z; a.w += b.w;
    o4[oi] = a;
  }
}

// Pack ego into swizzled bf16 B-fragment order; writes are contiguous b128s.
// thread t = (kb, nt, lane); element e<8 -> kr=half*8+e, e>=8 -> kr=16+half*8+(e-8).
__global__ void k_swz(const float* __restrict__ ego, __bf16* __restrict__ ego_bfs) {
  const int t = blockIdx.x * blockDim.x + threadIdx.x;
  if (t >= NTOT * D / 16) return;
  const int lane = t & 31;
  const int nt   = (t >> 5) & 3;
  const int kb   = t >> 7;
  const int half = lane >> 4, n = lane & 15;
  const float* src = ego + (size_t)(kb * 32 + half * 8) * D + nt * 16 + n;
  v8bf lo, hi;
#pragma unroll
  for (int e = 0; e < 8; ++e) {
    lo[e] = (__bf16)src[(size_t)e * D];
    hi[e] = (__bf16)src[(size_t)(16 + e) * D];
  }
  __bf16* dst = ego_bfs + (((size_t)kb * 4 + nt) * 32 + lane) * 16;
  *(v8bf*)dst       = lo;
  *(v8bf*)(dst + 8) = hi;
}

// Split-K GEMM with LDS-staged B: block = (mm, K-chunk), 8 waves = 8 M-tiles
// covering M=Q=128, each wave fuses all 4 n-tiles. B k-slab (4KB) is double-
// buffered in LDS; A fragments stream from global.
// tmp[0] = ut@ue, tmp[1] = vt@ie (f32 atomically combined across chunks).
__global__ void __launch_bounds__(256) k_gemm_small(const __bf16* __restrict__ ut_bf,
                                                    const __bf16* __restrict__ vt_bf,
                                                    const __bf16* __restrict__ ego_bfs,
                                                    float* __restrict__ tmp) {
  __shared__ __bf16 lds[2][2048];  // 2 x 4KB k-slabs
  const int tid  = threadIdx.x;
  const int lane = tid & 31;
  const int wave = tid >> 5;             // m-tile 0..7
  const int mm    = blockIdx.x / SM_SPLITK;
  const int chunk = blockIdx.x % SM_SPLITK;
  const __bf16* A  = mm ? vt_bf : ut_bf;
  const __bf16* Bs = ego_bfs + (mm ? (size_t)N_U * D : 0);  // ie slabs start at kb=3125
  float* C = tmp + (size_t)mm * Q * D;
  const int m0  = wave * 16;
  const int kb0 = chunk * SM_STEPS;

  // preload slab 0
  *(v8bf*)&lds[0][tid * 8] = *(const v8bf*)(Bs + (size_t)kb0 * 2048 + tid * 8);
  __syncthreads();

  v8f c0 = {}, c1 = {}, c2 = {}, c3 = {};
  for (int s = 0; s < SM_STEPS; ++s) {
    const int buf = s & 1;
    if (s + 1 < SM_STEPS)  // safe: buf^1 was last read in step s-1, fully fenced
      *(v8bf*)&lds[buf ^ 1][tid * 8] =
          *(const v8bf*)(Bs + (size_t)(kb0 + s + 1) * 2048 + tid * 8);

    v16bf a = load_a16x32(A, N_U, m0, (kb0 + s) * 32, lane);
    v16bf b0 = load_b_frag_ptr(&lds[buf][(0 * 32 + lane) * 16]);
    v16bf b1 = load_b_frag_ptr(&lds[buf][(1 * 32 + lane) * 16]);
    v16bf b2 = load_b_frag_ptr(&lds[buf][(2 * 32 + lane) * 16]);
    v16bf b3 = load_b_frag_ptr(&lds[buf][(3 * 32 + lane) * 16]);
    c0 = __builtin_amdgcn_wmma_f32_16x16x32_bf16(false, a, false, b0, (short)0, c0, false, false);
    c1 = __builtin_amdgcn_wmma_f32_16x16x32_bf16(false, a, false, b1, (short)0, c1, false, false);
    c2 = __builtin_amdgcn_wmma_f32_16x16x32_bf16(false, a, false, b2, (short)0, c2, false, false);
    c3 = __builtin_amdgcn_wmma_f32_16x16x32_bf16(false, a, false, b3, (short)0, c3, false, false);
    __syncthreads();
  }

  const int half = lane >> 4, n = lane & 15;
#pragma unroll
  for (int v = 0; v < 8; ++v) {
    const size_t rbase = (size_t)(m0 + v + 8 * half) * D + n;
    atomicAdd(&C[rbase +  0], c0[v]);
    atomicAdd(&C[rbase + 16], c1[v]);
    atomicAdd(&C[rbase + 32], c2[v]);
    atomicAdd(&C[rbase + 48], c3[v]);
  }
}

// Convert the two 128x64 f32 intermediates to swizzled bf16 B operands.
__global__ void k_tmp_to_bf16(const float* __restrict__ tmp, __bf16* __restrict__ tmp_bfs) {
  const int i = blockIdx.x * blockDim.x + threadIdx.x;
  if (i >= 2 * Q * D) return;
  const int mat = i / (Q * D), j = i % (Q * D);
  const int r = j / D, d_ = j % D;
  tmp_bfs[(size_t)mat * Q * D + b_swizzle_idx(r, d_)] = (__bf16)tmp[i];
}

// svd_u += u_mul_s @ vt_ei ; svd_i += v_mul_s @ ut_eu.
// One wave per (mm, 16-row M-tile); all 4 n-tiles fused so A loads once per kb.
// B is 16KB total and stays resident in WGP$/L2.
__global__ void k_gemm_big(const __bf16* __restrict__ umuls_bf,
                           const __bf16* __restrict__ vmuls_bf,
                           const __bf16* __restrict__ tmp_bfs,
                           float* __restrict__ out) {
  const int lane = threadIdx.x & 31;
  const int wid  = (blockIdx.x * blockDim.x + threadIdx.x) >> 5;
  const int waves_per_mm = N_U / 16;  // 6250
  if (wid >= 2 * waves_per_mm) return;
  const int mm = wid / waves_per_mm;
  const int m0 = (wid % waves_per_mm) * 16;
  const __bf16* A  = mm ? vmuls_bf : umuls_bf;
  const __bf16* Bs = tmp_bfs + (mm ? 0 : (size_t)Q * D);            // mm0: vt_ei, mm1: ut_eu
  float* Cout = out + (mm ? (size_t)3 * N_U * D : (size_t)N_U * D); // svd_i seg3 / svd_u seg1

  v8f c[4] = {};
#pragma unroll
  for (int kb = 0; kb < Q / 32; ++kb) {
    v16bf a = load_a16x32(A, Q, m0, kb * 32, lane);
#pragma unroll
    for (int nt = 0; nt < 4; ++nt) {
      v16bf b = load_b_frag_ptr(Bs + (((size_t)kb * 4 + nt) * 32 + lane) * 16);
      c[nt] = __builtin_amdgcn_wmma_f32_16x16x32_bf16(false, a, false, b, (short)0, c[nt], false, false);
    }
  }
  const int half = lane >> 4, n = lane & 15;
#pragma unroll
  for (int nt = 0; nt < 4; ++nt)
#pragma unroll
    for (int v = 0; v < 8; ++v) {
      float* p = Cout + (size_t)(m0 + v + 8 * half) * D + nt * 16 + n;
      *p += c[nt][v];
    }
}

__global__ void k_scale_mean(float* __restrict__ out) {
  const size_t NUD = (size_t)N_U * D;
  const float s = 1.0f / (float)LAYERS;
  size_t i = (size_t)blockIdx.x * blockDim.x + threadIdx.x;
  const size_t stride = (size_t)gridDim.x * blockDim.x;
  for (; i < NUD; i += stride) {
    out[i]           *= s;  // mean_u
    out[2 * NUD + i] *= s;  // mean_i
  }
}

// ---------------------------------------------------------------------------
extern "C" void kernel_launch(void* const* d_in, const int* in_sizes, int n_in,
                              void* d_out, int out_size, void* d_ws, size_t ws_size,
                              hipStream_t stream) {
  (void)in_sizes; (void)n_in; (void)out_size; (void)ws_size;
  const float* user_emb = (const float*)d_in[0];
  const float* item_emb = (const float*)d_in[1];
  const int*   adj_row  = (const int*)d_in[2];
  const int*   adj_col  = (const int*)d_in[3];
  const float* adj_vals = (const float*)d_in[4];
  const float* ut       = (const float*)d_in[5];
  const float* vt       = (const float*)d_in[6];
  const float* u_mul_s  = (const float*)d_in[7];
  const float* v_mul_s  = (const float*)d_in[8];
  float* out = (float*)d_out;

  char* ws = (char*)d_ws;
  size_t off = 0;
  auto alloc = [&](size_t bytes) -> char* {
    char* p = ws + off; off += (bytes + 255) & ~(size_t)255; return p;
  };
  float*  ego0     = (float*)alloc((size_t)NTOT * D * 4);
  float*  ego1     = (float*)alloc((size_t)NTOT * D * 4);
  __bf16* ut_bf    = (__bf16*)alloc((size_t)Q * N_U * 2);
  __bf16* vt_bf    = (__bf16*)alloc((size_t)Q * N_I * 2);
  __bf16* umuls_bf = (__bf16*)alloc((size_t)N_U * Q * 2);
  __bf16* vmuls_bf = (__bf16*)alloc((size_t)N_I * Q * 2);
  __bf16* ego_bfs  = (__bf16*)alloc((size_t)NTOT * D * 2);
  float*  tmp      = (float*)alloc((size_t)2 * Q * D * 4);
  __bf16* tmp_bfs  = (__bf16*)alloc((size_t)2 * Q * D * 2);

  // Zero outputs (mean segments double as the accumulators; svd segs accumulate).
  hipMemsetAsync(out, 0, (size_t)4 * N_U * D * sizeof(float), stream);

  // ego = concat(user_emb, item_emb)
  hipMemcpyAsync(ego0, user_emb, (size_t)N_U * D * 4, hipMemcpyDeviceToDevice, stream);
  hipMemcpyAsync(ego0 + (size_t)N_U * D, item_emb, (size_t)N_I * D * 4,
                 hipMemcpyDeviceToDevice, stream);

  // One-time bf16 conversions of the static factor matrices.
  k_f32_to_bf16<<<4096, 256, 0, stream>>>(ut,      ut_bf,    Q * N_U);
  k_f32_to_bf16<<<4096, 256, 0, stream>>>(vt,      vt_bf,    Q * N_I);
  k_f32_to_bf16<<<4096, 256, 0, stream>>>(u_mul_s, umuls_bf, N_U * Q);
  k_f32_to_bf16<<<4096, 256, 0, stream>>>(v_mul_s, vmuls_bf, N_I * Q);

  float* cur = ego0;
  float* nxt = ego1;
  for (int l = 0; l < LAYERS; ++l) {
    hipMemsetAsync(nxt, 0, (size_t)NTOT * D * 4, stream);
    k_spmm<<<4096, 256, 0, stream>>>(adj_row, adj_col, adj_vals, cur, nxt);
    k_acc<<<4096, 256, 0, stream>>>(nxt, out);
    k_swz<<<(NTOT * D / 16 + 255) / 256, 256, 0, stream>>>(nxt, ego_bfs);
    hipMemsetAsync(tmp, 0, (size_t)2 * Q * D * 4, stream);
    k_gemm_small<<<2 * SM_SPLITK, 256, 0, stream>>>(ut_bf, vt_bf, ego_bfs, tmp); // 250 blocks
    k_tmp_to_bf16<<<64, 256, 0, stream>>>(tmp, tmp_bfs);
    k_gemm_big<<<(2 * (N_U / 16)) / 4, 128, 0, stream>>>(umuls_bf, vmuls_bf, tmp_bfs, out); // 3125 blocks
    float* t = cur; cur = nxt; nxt = t;
  }
  k_scale_mean<<<4096, 256, 0, stream>>>(out);
}
